// GroupLocalAttention_49589692399774
// MI455X (gfx1250) — compile-verified, matrix operand order checked by reading, not run on hardware
//
#include <hip/hip_runtime.h>
#include <hip/hip_bf16.h>

#define B_    8
#define NMAX  8192
#define C_    384
#define G_    512
#define K_    32
#define H_    6

// Padded LDS strides (bank-conflict-free fragment access on 64x4B banks)
#define XSTRIDE 388   // halves: xg / q / k / attn rows (194 dwords, bank step 2)
#define VSTRIDE 36    // halves: vT / probs rows        (18 dwords, bank step 18)
#define SSTRIDE 33    // floats: score rows             (odd -> distinct banks)

typedef __attribute__((ext_vector_type(16))) __bf16 v16bf;
typedef __attribute__((ext_vector_type(8)))  float  v8f;

union FragAB {
    v16bf v;
    unsigned short s[16];
    unsigned int   u[8];
};

__device__ __forceinline__ unsigned short f32_to_bf16(float f) {
    union { float f; unsigned int u; } cv; cv.f = f;
    unsigned int u = cv.u;
    u += 0x7FFFu + ((u >> 16) & 1u);   // round-to-nearest-even
    return (unsigned short)(u >> 16);
}

// ISA 7.12.2 16-bit A/B fragment: element e of lane (lo/hi half) -> K index
__device__ __forceinline__ int kmap(int e, int hi) {
    return e + (e < 8 ? (hi ? 8 : 0) : (hi ? 16 : 8));
}

// Per-lane fragment load from a row-major [N-or-M][K] bf16 LDS buffer.
// Works for A fragments and for B fragments when B^T is stored row-major,
// because both stripe one row per lane with the same K-pair mapping.
// rp = buffer + lane_row * stride + k_tile_base ; pairs are 32-bit aligned.
__device__ __forceinline__ void load_frag_rows(FragAB& f,
                                               const unsigned short* rp,
                                               int hi) {
#pragma unroll
    for (int j = 0; j < 8; ++j) {
        int k0 = (j < 4 ? 2 * j : 8 + 2 * j) + (hi ? 8 : 0);
        f.u[j] = *(const unsigned int*)(rp + k0);
    }
}

// ---------------------------------------------------------------------------
// Kernel 1: swizzle fp32 weights into bf16 WMMA B-fragments.
// B[k][n] = W[n_global][k_global]  (because out = x * W^T)
// Fragment f = nt*12 + kt holds 32 lanes x 16 bf16 contiguous (one v16bf/lane).
// ---------------------------------------------------------------------------
__global__ void prep_weights_kernel(const float* __restrict__ wqkv,
                                    const float* __restrict__ wproj,
                                    unsigned short* __restrict__ fqkv,
                                    unsigned short* __restrict__ fproj) {
    int t = blockIdx.x * 256 + threadIdx.x;
    const int NQ = 72 * 12 * 512;   // 1152/16 n-tiles * 384/32 k-tiles * 512
    const int NP = 24 * 12 * 512;   // 384/16  n-tiles * 384/32 k-tiles * 512
    if (t < NQ) {
        int f = t >> 9, r = t & 511, lane = r >> 4, e = r & 15;
        int nt = f / 12, kt = f % 12;
        int n  = nt * 16 + (lane & 15);
        int kg = kt * 32 + kmap(e, lane >> 4);
        fqkv[t] = f32_to_bf16(wqkv[n * C_ + kg]);
    } else if (t < NQ + NP) {
        int u = t - NQ;
        int f = u >> 9, r = u & 511, lane = r >> 4, e = r & 15;
        int nt = f / 12, kt = f % 12;
        int n  = nt * 16 + (lane & 15);
        int kg = kt * 32 + kmap(e, lane >> 4);
        fproj[u] = f32_to_bf16(wproj[n * C_ + kg]);
    }
}

// ---------------------------------------------------------------------------
// Kernel 2: one workgroup (8 wave32 = 256 thr) per group (4096 groups).
// Dynamic LDS layout (116736 B):
//   [0     , 25344) xg bf16[32][388]     -> reused as scores f32[6][32][33]
//   [25344 , 50176) q  bf16[32][388]   (scale 1/8 folded in)
//   [50176 , 75008) k  bf16[32][388]     -> reused as attn_out bf16[32][388]
//   [75008 ,102656) vT bf16[384][36]   (v stored transposed: [d][key])
//   [102656,116480) probs bf16[6][32][36]
//   [116480,116608) mask f32[32]
//   [116608,116736) pidx i32[32]
// ---------------------------------------------------------------------------
__global__ __launch_bounds__(256)
void gla_attention_kernel(const float*          __restrict__ feats,
                          const long long*      __restrict__ gidx,
                          const float*          __restrict__ gmask,
                          const unsigned short* __restrict__ fqkv,
                          const unsigned short* __restrict__ fproj,
                          const float*          __restrict__ bproj,
                          float*                __restrict__ acc_out,
                          float*                __restrict__ cnt) {
    extern __shared__ char smem[];
    unsigned short* xg     = (unsigned short*)smem;
    float*          scores = (float*)smem;
    unsigned short* qs     = (unsigned short*)(smem + 25344);
    unsigned short* ks     = (unsigned short*)(smem + 50176);
    unsigned short* attn   = ks;
    unsigned short* vT     = (unsigned short*)(smem + 75008);
    unsigned short* probs  = (unsigned short*)(smem + 102656);
    float*          maskS  = (float*)(smem + 116480);
    int*            pidx   = (int*)(smem + 116608);

    const int tid  = threadIdx.x;
    const int lane = tid & 31;
    const int wave = tid >> 5;
    const int bg   = blockIdx.x;
    const int b    = bg / G_;
    const int nl   = lane & 15;
    const int hi   = lane >> 4;
    const int hoff = hi ? 8 : 0;

    // --- indices, mask, count scatter ---
    if (tid < K_) {
        long long raw = gidx[(size_t)bg * K_ + tid];
        int p = raw < 0 ? 0 : (int)raw;
        pidx[tid] = p;
        float m = gmask[(size_t)bg * K_ + tid];
        maskS[tid] = m;
        __hip_atomic_fetch_add(&cnt[(size_t)b * NMAX + p], m,
                               __ATOMIC_RELAXED, __HIP_MEMORY_SCOPE_AGENT);
    }
    __syncthreads();

    // --- gather x -> LDS bf16 (float4 loads, packed b64 stores) ---
    for (int ch = tid; ch < K_ * (C_ / 4); ch += 256) {
        int row = ch / (C_ / 4), c4 = (ch - row * (C_ / 4)) * 4;
        const float4 f = *(const float4*)(feats +
            ((size_t)b * NMAX + pidx[row]) * C_ + c4);
        unsigned long long pk =
              (unsigned long long)f32_to_bf16(f.x)
            | ((unsigned long long)f32_to_bf16(f.y) << 16)
            | ((unsigned long long)f32_to_bf16(f.z) << 32)
            | ((unsigned long long)f32_to_bf16(f.w) << 48);
        *(unsigned long long*)(xg + row * XSTRIDE + c4) = pk;
    }
    __syncthreads();

    // --- QKV GEMM: [32x384] x [384x1152] ; 144 tiles x 12 WMMA steps ---
    for (int t = wave; t < 144; t += 8) {
        int mt = t / 72, nt = t % 72;
        v8f acc = {};
        for (int kt = 0; kt < 12; ++kt) {
            FragAB a;
            load_frag_rows(a, xg + (mt * 16 + nl) * XSTRIDE + kt * 32, hi);
            FragAB bf;
            bf.v = *((const v16bf*)fqkv + (size_t)(nt * 12 + kt) * 32 + lane);
            acc = __builtin_amdgcn_wmma_f32_16x16x32_bf16(
                false, a.v, false, bf.v, (short)0, acc, false, false);
        }
        int ng = nt * 16 + nl;
#pragma unroll
        for (int r = 0; r < 8; ++r) {
            int row = mt * 16 + hoff + r;
            float val = acc[r];
            if (ng < 384)       qs[row * XSTRIDE + ng] = f32_to_bf16(val * 0.125f);
            else if (ng < 768)  ks[row * XSTRIDE + (ng - 384)] = f32_to_bf16(val);
            else                vT[(ng - 768) * VSTRIDE + row] = f32_to_bf16(val);
        }
    }
    __syncthreads();

    // --- scores = q @ k^T per head: 24 tiles x 2 WMMA steps.
    //     B^T = k stored row-major [key][d] -> B frags load like A frags. ---
    for (int t = wave; t < 24; t += 8) {
        int h = t >> 2, mt = (t >> 1) & 1, nt = t & 1;
        int key = nt * 16 + nl;
        v8f acc = {};
        for (int s = 0; s < 2; ++s) {
            FragAB a, bf;
            load_frag_rows(a, qs + (mt * 16 + nl) * XSTRIDE + h * 64 + s * 32, hi);
            load_frag_rows(bf, ks + key * XSTRIDE + h * 64 + s * 32, hi);
            acc = __builtin_amdgcn_wmma_f32_16x16x32_bf16(
                false, a.v, false, bf.v, (short)0, acc, false, false);
        }
#pragma unroll
        for (int r = 0; r < 8; ++r)
            scores[(h * 32 + mt * 16 + hoff + r) * SSTRIDE + key] = acc[r];
    }
    __syncthreads();

    // --- masked softmax over keys, one row per thread (192 rows) ---
    if (tid < H_ * 32) {
        int h = tid >> 5, q = tid & 31;
        const float* srow = scores + (h * 32 + q) * SSTRIDE;
        float sb[32];
        float mx = -1e30f;
#pragma unroll
        for (int k = 0; k < 32; ++k) {
            float s = srow[k] + (maskS[k] > 0.f ? 0.f : -1e9f);
            sb[k] = s;
            mx = fmaxf(mx, s);
        }
        float sum = 0.f;
#pragma unroll
        for (int k = 0; k < 32; ++k) { float e = __expf(sb[k] - mx); sb[k] = e; sum += e; }
        float inv = 1.f / sum;
        unsigned short* prow = probs + (h * 32 + q) * VSTRIDE;
#pragma unroll
        for (int k = 0; k < 32; ++k) prow[k] = f32_to_bf16(sb[k] * inv);
    }
    __syncthreads();

    // --- out = attn @ v per head: 48 tiles x 1 WMMA (K=32).
    //     B^T = vT stored row-major [d][key] -> B frags load like A frags. ---
    for (int t = wave; t < 48; t += 8) {
        int h = t >> 3, rem = t & 7, mt = rem >> 2, nt = rem & 3;
        int col = h * 64 + nt * 16 + nl;
        FragAB a, bf;
        load_frag_rows(a, probs + (h * 32 + mt * 16 + nl) * VSTRIDE, hi);
        load_frag_rows(bf, vT + col * VSTRIDE, hi);
        v8f acc = {};
        acc = __builtin_amdgcn_wmma_f32_16x16x32_bf16(
            false, a.v, false, bf.v, (short)0, acc, false, false);
#pragma unroll
        for (int r = 0; r < 8; ++r)
            attn[(mt * 16 + hoff + r) * XSTRIDE + col] = f32_to_bf16(acc[r]);
    }
    __syncthreads();

    // --- proj GEMM [32x384]x[384x384] + bias, masked atomic scatter ---
    for (int t = wave; t < 48; t += 8) {
        int mt = t / 24, nt = t % 24;
        v8f acc = {};
        for (int kt = 0; kt < 12; ++kt) {
            FragAB a;
            load_frag_rows(a, attn + (mt * 16 + nl) * XSTRIDE + kt * 32, hi);
            FragAB bf;
            bf.v = *((const v16bf*)fproj + (size_t)(nt * 12 + kt) * 32 + lane);
            acc = __builtin_amdgcn_wmma_f32_16x16x32_bf16(
                false, a.v, false, bf.v, (short)0, acc, false, false);
        }
        int col = nt * 16 + nl;
        float bias = bproj[col];
#pragma unroll
        for (int r = 0; r < 8; ++r) {
            int row = mt * 16 + hoff + r;
            float val = (acc[r] + bias) * maskS[row];
            int p = pidx[row];
            __hip_atomic_fetch_add(&acc_out[((size_t)b * NMAX + p) * C_ + col], val,
                                   __ATOMIC_RELAXED, __HIP_MEMORY_SCOPE_AGENT);
        }
    }
}

// ---------------------------------------------------------------------------
// Kernel 3: out = feats + (acc / max(cnt,1)) * gamma
// ---------------------------------------------------------------------------
__global__ void gla_finalize_kernel(const float* __restrict__ feats,
                                    const float* __restrict__ cnt,
                                    const float* __restrict__ gamma,
                                    float* __restrict__ out) {
    size_t t = (size_t)blockIdx.x * 256 + threadIdx.x;
    if (t >= (size_t)B_ * NMAX * C_) return;
    size_t pn = t / C_;
    int c = (int)(t - pn * C_);
    float a  = out[t];
    float cc = cnt[pn];
    cc = cc > 1.f ? cc : 1.f;
    out[t] = feats[t] + (a / cc) * gamma[c];
}

extern "C" void kernel_launch(void* const* d_in, const int* in_sizes, int n_in,
                              void* d_out, int out_size, void* d_ws, size_t ws_size,
                              hipStream_t stream) {
    const float*     feats = (const float*)d_in[0];
    const long long* gidx  = (const long long*)d_in[1];   // int64 in reference
    const float*     gmask = (const float*)d_in[2];
    const float*     wqkv  = (const float*)d_in[3];
    const float*     wproj = (const float*)d_in[4];
    const float*     bproj = (const float*)d_in[5];
    const float*     gamma = (const float*)d_in[6];
    float* out = (float*)d_out;

    // workspace layout
    float*          cnt   = (float*)d_ws;                                     // 262144 B
    unsigned short* fqkv  = (unsigned short*)((char*)d_ws + 262144);          // 884736 B
    unsigned short* fproj = (unsigned short*)((char*)d_ws + 262144 + 884736); // 294912 B

    // d_out doubles as the scatter accumulator -> zero it (and counts) each call
    hipMemsetAsync(d_out, 0, (size_t)out_size * sizeof(float), stream);
    hipMemsetAsync(cnt, 0, (size_t)B_ * NMAX * sizeof(float), stream);

    const int prepTot = (72 * 12 + 24 * 12) * 512;
    prep_weights_kernel<<<(prepTot + 255) / 256, 256, 0, stream>>>(
        wqkv, wproj, fqkv, fproj);

    gla_attention_kernel<<<B_ * G_, 256, 116736, stream>>>(
        feats, gidx, gmask, fqkv, fproj, bproj, out, cnt);

    size_t tot = (size_t)B_ * NMAX * C_;
    gla_finalize_kernel<<<(int)((tot + 255) / 256), 256, 0, stream>>>(
        feats, cnt, gamma, out);
}
